// MultiSensor_HARMamba_v1_31851477467336
// MI455X (gfx1250) — matrix operand order repaired and checked by reading, not compiled, not run on hardware
//
#include <hip/hip_runtime.h>
#include <hip/hip_bf16.h>

// ---------------------------------------------------------------------------
// HARMamba forward for MI455X (gfx1250, wave32).
//  - All large GEMMs use v_wmma_f32_16x16x32_bf16 (bf16 in, f32 accum).
//    Each wave computes a 16x(16*NT) strip: A fragment loaded once per
//    k-step, reused across NT wmma issues (raises flop/byte 4 -> 6.4).
//  - SSM scan: one thread per (batch, d_inner) channel, 16-state in registers.
//  - ~60MB workspace, fits in 192MB L2; all kernels launched on `stream`.
// ---------------------------------------------------------------------------

#define BDIM   16
#define LSEQ   512
#define NP     32
#define TTOK   352
#define DM     128
#define DI     256
#define DSTATE 16
#define DCONV  4
#define DTR    8
#define DEPTHL 12
#define NCLS   18
#define EPSF   1e-5f

typedef __attribute__((ext_vector_type(16))) __bf16 v16bf;
typedef __attribute__((ext_vector_type(8)))  __bf16 v8bf;
typedef __attribute__((ext_vector_type(8)))  float  v8f;

// ------------------------- small utility kernels ---------------------------

__global__ void k_f2bf(const float* __restrict__ in, __bf16* __restrict__ out, int n) {
  int i = blockIdx.x * blockDim.x + threadIdx.x;
  if (i < n) out[i] = (__bf16)in[i];
}

__global__ void k_negexp(const float* __restrict__ in, float* __restrict__ out, int n) {
  int i = blockIdx.x * blockDim.x + threadIdx.x;
  if (i < n) out[i] = -__expf(in[i]);
}

__global__ void k_add(float* __restrict__ a, const float* __restrict__ b, int n) {
  int i = blockIdx.x * blockDim.x + threadIdx.x;
  if (i < n) a[i] += b[i];
}

// pad x_proj weight [12,40,256] -> bf16 [12,48,256] (rows 40..47 zero)
__global__ void k_pad_xproj(const float* __restrict__ w, __bf16* __restrict__ out) {
  int i = blockIdx.x * blockDim.x + threadIdx.x;
  if (i >= DEPTHL * 48 * DI) return;
  int k = i % DI;
  int n = (i / DI) % 48;
  int l = i / (48 * DI);
  out[i] = (n < 40) ? (__bf16)w[((size_t)l * 40 + n) * DI + k] : (__bf16)0.0f;
}

// reorder tof patch-embed weight [128,64,16] (d,c,k) -> bf16 [128, k*64+c]
__global__ void k_tofpe(const float* __restrict__ w, __bf16* __restrict__ out) {
  int i = blockIdx.x * blockDim.x + threadIdx.x;
  if (i >= DM * 1024) return;
  int kc = i % 1024, d = i / 1024;
  int k = kc / 64, c = kc % 64;
  out[i] = (__bf16)w[((size_t)d * 64 + c) * 16 + k];
}

// ------------------------------- RevIN --------------------------------------
// instance norm over time (512) per (instance, channel), affine.
__global__ void k_revin(const float* __restrict__ x, const float* __restrict__ w,
                        const float* __restrict__ b, float* __restrict__ out,
                        int inst, int C) {
  int i = blockIdx.x * blockDim.x + threadIdx.x;
  if (i >= inst * C) return;
  int c = i % C, id = i / C;
  const float* xp = x + (size_t)id * LSEQ * C + c;
  float s = 0.f;
  for (int t = 0; t < LSEQ; ++t) s += xp[(size_t)t * C];
  float m = s / (float)LSEQ;
  float v = 0.f;
  for (int t = 0; t < LSEQ; ++t) { float d0 = xp[(size_t)t * C] - m; v += d0 * d0; }
  v /= (float)LSEQ;
  float inv = rsqrtf(v + EPSF);
  float wc = w[c], bc = b[c];
  float* op = out + (size_t)id * LSEQ * C + c;
  for (int t = 0; t < LSEQ; ++t) op[(size_t)t * C] = (xp[(size_t)t * C] - m) * inv * wc + bc;
}

// ----------------------- scalar patch embed (imu/thm) ----------------------
__global__ void k_pembed(const float* __restrict__ xr, const float* __restrict__ w,
                         float* __restrict__ out, int C, int instances) {
  int i = blockIdx.x * blockDim.x + threadIdx.x;
  int total = instances * NP * DM;
  if (i >= total) return;
  int d = i % DM;
  int p = (i / DM) % NP;
  int inst = i / (DM * NP);
  const float* xp = xr + ((size_t)inst * LSEQ + p * 16) * C;  // [16,C] contiguous
  const float* wp = w + (size_t)d * C * 16;                   // [C,16]
  float acc = 0.f;
  for (int c = 0; c < C; ++c)
    for (int k = 0; k < 16; ++k)
      acc += xp[k * C + c] * wp[c * 16 + k];
  out[i] = acc;
}

// --------------------------- WMMA bf16 GEMM --------------------------------
// C[M,N](f32,ldc) = A[M,K](bf16,lda) @ W[N,K]^T(bf16,ldb) (+bias[n])
// One wave computes a 16 x (16*NT) strip; A fragment reused across NT tiles.
// Requires M%64==0, N%(16*NT)==0, K%32==0.
template <int NT>
__global__ void k_gemm_wmma(const __bf16* __restrict__ A, int lda,
                            const __bf16* __restrict__ W, int ldb,
                            float* __restrict__ C, int ldc,
                            const float* __restrict__ bias,
                            int M, int N, int K) {
  int n0 = blockIdx.x * (16 * NT);
  int m0 = (blockIdx.y * blockDim.y + threadIdx.y) * 16;
  if (m0 >= M || n0 >= N) return;          // wave-uniform: EXEC stays all-ones
  int lane = threadIdx.x & 31;
  int row = lane & 15;
  int hi  = lane >> 4;
  const __bf16* arow = A + (size_t)(m0 + row) * lda;
  const __bf16* wrow[NT];
#pragma unroll
  for (int nt = 0; nt < NT; ++nt)
    wrow[nt] = W + (size_t)(n0 + nt * 16 + row) * ldb;
  v8f acc[NT];
#pragma unroll
  for (int nt = 0; nt < NT; ++nt) acc[nt] = (v8f){};
  for (int k0 = 0; k0 < K; k0 += 32) {
    // A fragment: lane-half 0 holds K 0..7 & 16..23, half 1 holds 8..15 & 24..31
    v8bf a0 = *(const v8bf*)(arow + k0 + hi * 8);
    v8bf a1 = *(const v8bf*)(arow + k0 + 16 + hi * 8);
    v16bf av;
#pragma unroll
    for (int j = 0; j < 8; ++j) { av[j] = a0[j]; av[j + 8] = a1[j]; }
#pragma unroll
    for (int nt = 0; nt < NT; ++nt) {
      // B fragment: lane-half 0 holds K 0..7 & 16..23, half 1 holds 8..15 & 24..31 (one column)
      v8bf b0 = *(const v8bf*)(wrow[nt] + k0 + hi * 8);
      v8bf b1 = *(const v8bf*)(wrow[nt] + k0 + 16 + hi * 8);
      v16bf bv;
#pragma unroll
      for (int j = 0; j < 8; ++j) { bv[j] = b0[j]; bv[j + 8] = b1[j]; }
      acc[nt] = __builtin_amdgcn_wmma_f32_16x16x32_bf16(
          false, av, false, bv, (short)0, acc[nt], false, false);
    }
  }
#pragma unroll
  for (int nt = 0; nt < NT; ++nt) {
    int nc = n0 + nt * 16 + row;
    float bb = bias ? bias[nc] : 0.0f;
    float* cc = C + (size_t)(m0 + hi * 8) * ldc + nc;
#pragma unroll
    for (int v = 0; v < 8; ++v) cc[(size_t)v * ldc] = acc[nt][v] + bb;
  }
}

// ----------------------------- assemble tokens -----------------------------
__global__ void k_assemble(const float* __restrict__ timu, const float* __restrict__ ttof,
                           const float* __restrict__ tthm, const float* __restrict__ pbimu,
                           const float* __restrict__ pbtof, const float* __restrict__ pbthm,
                           const float* __restrict__ ste, const float* __restrict__ pos,
                           float* __restrict__ res) {
  int i = blockIdx.x * blockDim.x + threadIdx.x;
  if (i >= BDIM * TTOK * DM) return;
  int d = i % DM;
  int j = (i / DM) % TTOK;
  int b = i / (DM * TTOK);
  float v, pb, se;
  if (j < NP) {
    v = timu[((size_t)b * NP + j) * DM + d]; pb = pbimu[d]; se = ste[0 * DM + d];
  } else if (j < 6 * NP) {
    int jj = j - NP; int s = jj >> 5, p = jj & 31;
    v = ttof[(((size_t)b * 5 + s) * NP + p) * DM + d]; pb = pbtof[d]; se = ste[1 * DM + d];
  } else {
    int jj = j - 6 * NP; int s = jj >> 5, p = jj & 31;
    v = tthm[(((size_t)b * 5 + s) * NP + p) * DM + d]; pb = pbthm[d]; se = ste[2 * DM + d];
  }
  res[i] = v + pb + se + pos[(size_t)j * DM + d];
}

// ------------------------------- LayerNorm ---------------------------------
// one block (128 threads) per row
__global__ void k_ln(const float* __restrict__ x, const float* __restrict__ w,
                     const float* __restrict__ b, __bf16* __restrict__ obf,
                     float* __restrict__ of) {
  int r = blockIdx.x;
  int d = threadIdx.x;
  __shared__ float red[DM];
  float v = x[(size_t)r * DM + d];
  red[d] = v; __syncthreads();
  for (int s = DM / 2; s > 0; s >>= 1) { if (d < s) red[d] += red[d + s]; __syncthreads(); }
  float m = red[0] / (float)DM; __syncthreads();
  float dx = v - m;
  red[d] = dx * dx; __syncthreads();
  for (int s = DM / 2; s > 0; s >>= 1) { if (d < s) red[d] += red[d + s]; __syncthreads(); }
  float var = red[0] / (float)DM;
  float y = dx * rsqrtf(var + EPSF) * w[d] + b[d];
  if (obf) obf[(size_t)r * DM + d] = (__bf16)y;
  if (of)  of[(size_t)r * DM + d] = y;
}

// ---------------- causal depthwise conv (k=4) + SiLU, direction-aware -------
__global__ void k_conv_silu(const float* __restrict__ xz, const float* __restrict__ cw,
                            const float* __restrict__ cb, float* __restrict__ xc,
                            __bf16* __restrict__ xcbf, int dir) {
  int i = blockIdx.x * blockDim.x + threadIdx.x;
  if (i >= BDIM * TTOK * DI) return;
  int di = i % DI;
  int t = (i / DI) % TTOK;
  int b = i / (DI * TTOK);
  float acc = cb[di];
#pragma unroll
  for (int k = 0; k < DCONV; ++k) {
    int tt = t - (DCONV - 1) + k;
    if (tt >= 0) {
      int ts = dir ? (TTOK - 1 - tt) : tt;
      acc += xz[(((size_t)b * TTOK + ts) * 2 * DI) + di] * cw[di * DCONV + k];
    }
  }
  float s = acc / (1.f + __expf(-acc));
  xc[i] = s;
  xcbf[i] = (__bf16)s;
}

// ------------------------- dt projection + softplus ------------------------
__global__ void k_dt(const float* __restrict__ dbc, const float* __restrict__ dtw,
                     const float* __restrict__ dtb, float* __restrict__ delta) {
  int i = blockIdx.x * blockDim.x + threadIdx.x;
  if (i >= BDIM * TTOK * DI) return;
  int di = i % DI;
  size_t bt = (size_t)(i / DI);
  const float* dr = dbc + bt * 48;
  float acc = dtb[di];
#pragma unroll
  for (int r = 0; r < DTR; ++r) acc += dr[r] * dtw[di * DTR + r];
  delta[i] = (acc > 20.f) ? acc : log1pf(__expf(acc));
}

// --------------------------- selective-scan SSM ----------------------------
// one thread per (b, d_inner); 16-state recurrence in registers; gated output
__global__ void k_scan(const float* __restrict__ dbc, const float* __restrict__ delta,
                       const float* __restrict__ xc, const float* __restrict__ xz,
                       const float* __restrict__ A, const float* __restrict__ Dp,
                       float* __restrict__ yout, int dir) {
  int i = blockIdx.x * blockDim.x + threadIdx.x;
  if (i >= BDIM * DI) return;
  int di = i % DI;
  int b = i / DI;
  float h[DSTATE], a_[DSTATE];
#pragma unroll
  for (int n = 0; n < DSTATE; ++n) { h[n] = 0.f; a_[n] = A[di * DSTATE + n]; }
  float dp = Dp[di];
  for (int t = 0; t < TTOK; ++t) {
    size_t bt = (size_t)b * TTOK + t;
    float dl = delta[bt * DI + di];
    float u  = xc[bt * DI + di];
    const float* bc = dbc + bt * 48 + DTR;
    const float* cc = bc + DSTATE;
    float du = dl * u;
    float y = 0.f;
#pragma unroll
    for (int n = 0; n < DSTATE; ++n) {
      h[n] = __expf(dl * a_[n]) * h[n] + du * bc[n];
      y += h[n] * cc[n];
    }
    y += u * dp;
    int tout = dir ? (TTOK - 1 - t) : t;
    float z = xz[(((size_t)b * TTOK + tout) * 2 * DI) + DI + di];
    float g = z / (1.f + __expf(-z));
    float val = y * g;
    size_t oi = ((size_t)b * TTOK + tout) * DI + di;
    if (dir) yout[oi] += val; else yout[oi] = val;
  }
}

// ------------------------------- mean + head -------------------------------
__global__ void k_meanT(const float* __restrict__ x, float* __restrict__ out) {
  int i = blockIdx.x * blockDim.x + threadIdx.x;
  if (i >= BDIM * DM) return;
  int d = i % DM, b = i / DM;
  float s = 0.f;
  for (int t = 0; t < TTOK; ++t) s += x[((size_t)b * TTOK + t) * DM + d];
  out[i] = s / (float)TTOK;
}

__global__ void k_head(const float* __restrict__ mv, const float* __restrict__ hw,
                       const float* __restrict__ hb, float* __restrict__ out) {
  int i = blockIdx.x * blockDim.x + threadIdx.x;
  if (i >= BDIM * NCLS) return;
  int c = i % NCLS, b = i / NCLS;
  float s = hb[c];
  for (int d = 0; d < DM; ++d) s += mv[b * DM + d] * hw[c * DM + d];
  out[(size_t)b * NCLS + c] = s;
}

// ------------------------------ host launcher ------------------------------

static inline int cdiv(int a, int b) { return (a + b - 1) / b; }

extern "C" void kernel_launch(void* const* d_in, const int* in_sizes, int n_in,
                              void* d_out, int out_size, void* d_ws, size_t ws_size,
                              hipStream_t stream) {
  (void)in_sizes; (void)n_in; (void)out_size; (void)ws_size;
  const float* imu    = (const float*)d_in[0];
  const float* thm    = (const float*)d_in[1];
  const float* tof    = (const float*)d_in[2];
  const float* rw_imu = (const float*)d_in[3];
  const float* rb_imu = (const float*)d_in[4];
  const float* rw_tof = (const float*)d_in[5];
  const float* rb_tof = (const float*)d_in[6];
  const float* rw_thm = (const float*)d_in[7];
  const float* rb_thm = (const float*)d_in[8];
  const float* pew_imu= (const float*)d_in[9];
  const float* peb_imu= (const float*)d_in[10];
  const float* pew_tof= (const float*)d_in[11];
  const float* peb_tof= (const float*)d_in[12];
  const float* pew_thm= (const float*)d_in[13];
  const float* peb_thm= (const float*)d_in[14];
  const float* ste    = (const float*)d_in[15];
  const float* pos    = (const float*)d_in[16];
  const float* norm_w = (const float*)d_in[17];
  const float* norm_b = (const float*)d_in[18];
  const float* inpw   = (const float*)d_in[19];
  const float* convw  = (const float*)d_in[20];
  const float* convb  = (const float*)d_in[21];
  const float* xprojw = (const float*)d_in[22];
  const float* dtw    = (const float*)d_in[23];
  const float* dtb    = (const float*)d_in[24];
  const float* Alog   = (const float*)d_in[25];
  const float* Ablog  = (const float*)d_in[26];
  const float* Dparam = (const float*)d_in[27];
  const float* outpw  = (const float*)d_in[28];
  const float* nfw    = (const float*)d_in[29];
  const float* nfb    = (const float*)d_in[30];
  const float* headw  = (const float*)d_in[31];
  const float* headb  = (const float*)d_in[32];
  float* out = (float*)d_out;

  char* wsb = (char*)d_ws;
  size_t off = 0;
  auto alloc = [&](size_t bytes) -> void* {
    off = (off + 255) & ~(size_t)255;
    void* p = wsb + off;
    off += bytes;
    return p;
  };

  const int N_RES = BDIM * TTOK * DM;       // 720896
  const int N_XZ  = BDIM * TTOK * 2 * DI;   // 2883584
  const int N_XC  = BDIM * TTOK * DI;       // 1441792
  const int N_DBC = BDIM * TTOK * 48;       // 270336

  float*  res    = (float*) alloc((size_t)N_RES * 4);
  __bf16* hnbf   = (__bf16*)alloc((size_t)N_RES * 2);
  float*  xz     = (float*) alloc((size_t)N_XZ * 4);
  float*  xc     = (float*) alloc((size_t)N_XC * 4);
  __bf16* xcbf   = (__bf16*)alloc((size_t)N_XC * 2);
  float*  dbc    = (float*) alloc((size_t)N_DBC * 4);
  float*  delta  = (float*) alloc((size_t)N_XC * 4);
  float*  ybuf   = (float*) alloc((size_t)N_XC * 4);
  __bf16* ybf    = (__bf16*)alloc((size_t)N_XC * 2);
  float*  hid    = (float*) alloc((size_t)N_RES * 4);
  __bf16* w_in   = (__bf16*)alloc((size_t)DEPTHL * 512 * DM * 2);
  __bf16* w_xp   = (__bf16*)alloc((size_t)DEPTHL * 48 * DI * 2);
  __bf16* w_op   = (__bf16*)alloc((size_t)DEPTHL * DM * DI * 2);
  __bf16* w_tpe  = (__bf16*)alloc((size_t)DM * 1024 * 2);
  float*  negA   = (float*) alloc((size_t)DEPTHL * DI * DSTATE * 4);
  float*  negAb  = (float*) alloc((size_t)DEPTHL * DI * DSTATE * 4);
  float*  rimu   = (float*) alloc((size_t)BDIM * LSEQ * 7 * 4);
  float*  rtof   = (float*) alloc((size_t)80 * LSEQ * 64 * 4);
  __bf16* rtofbf = (__bf16*)alloc((size_t)80 * LSEQ * 64 * 2);
  float*  rthm   = (float*) alloc((size_t)80 * LSEQ * 4);
  float*  t_imu  = (float*) alloc((size_t)BDIM * NP * DM * 4);
  float*  t_tof  = (float*) alloc((size_t)80 * NP * DM * 4);
  float*  t_thm  = (float*) alloc((size_t)80 * NP * DM * 4);
  float*  meanv  = (float*) alloc((size_t)BDIM * DM * 4);

  const int TB = 256;
  const dim3 gblk(32, 4);

  // ---- weight preparation (bf16 conversions, deterministic every call) ----
  { int n = DEPTHL * 512 * DM; k_f2bf<<<cdiv(n, TB), TB, 0, stream>>>(inpw, w_in, n); }
  { int n = DEPTHL * DM * DI;  k_f2bf<<<cdiv(n, TB), TB, 0, stream>>>(outpw, w_op, n); }
  { int n = DEPTHL * 48 * DI;  k_pad_xproj<<<cdiv(n, TB), TB, 0, stream>>>(xprojw, w_xp); }
  { int n = DM * 1024;         k_tofpe<<<cdiv(n, TB), TB, 0, stream>>>(pew_tof, w_tpe); }
  { int n = DEPTHL * DI * DSTATE;
    k_negexp<<<cdiv(n, TB), TB, 0, stream>>>(Alog, negA, n);
    k_negexp<<<cdiv(n, TB), TB, 0, stream>>>(Ablog, negAb, n); }

  // ---- RevIN ----
  k_revin<<<cdiv(BDIM * 7, TB), TB, 0, stream>>>(imu, rw_imu, rb_imu, rimu, BDIM, 7);
  k_revin<<<cdiv(80 * 64, TB), TB, 0, stream>>>(tof, rw_tof, rb_tof, rtof, 80, 64);
  k_revin<<<cdiv(80 * 1, TB), TB, 0, stream>>>(thm, rw_thm, rb_thm, rthm, 80, 1);
  { int n = 80 * LSEQ * 64; k_f2bf<<<cdiv(n, TB), TB, 0, stream>>>(rtof, rtofbf, n); }

  // ---- patch embeds ----
  k_pembed<<<cdiv(BDIM * NP * DM, TB), TB, 0, stream>>>(rimu, pew_imu, t_imu, 7, BDIM);
  k_pembed<<<cdiv(80 * NP * DM, TB), TB, 0, stream>>>(rthm, pew_thm, t_thm, 1, 80);
  // tof: WMMA GEMM  [2560,1024] x [128,1024]^T   (N=128 -> 2 strips of 64)
  k_gemm_wmma<4><<<dim3(DM / 64, 2560 / 64), gblk, 0, stream>>>(
      rtofbf, 1024, w_tpe, 1024, t_tof, DM, nullptr, 2560, DM, 1024);

  // ---- assemble residual (embeds + biases + sensor + pos) ----
  k_assemble<<<cdiv(N_RES, TB), TB, 0, stream>>>(
      t_imu, t_tof, t_thm, peb_imu, peb_tof, peb_thm, ste, pos, res);

  const int M = BDIM * TTOK;  // 5632, multiple of 64
  for (int i = 0; i < DEPTHL; ++i) {
    // LN -> bf16
    k_ln<<<M, DM, 0, stream>>>(res, norm_w + i * DM, norm_b + i * DM, hnbf, (float*)nullptr);
    // in_proj: [5632,128] x [512,128]^T -> xz   (N=512 -> 8 strips of 64)
    k_gemm_wmma<4><<<dim3(512 / 64, M / 64), gblk, 0, stream>>>(
        hnbf, DM, w_in + (size_t)i * 512 * DM, DM, xz, 2 * DI, nullptr, M, 512, DM);

    for (int dir = 0; dir < 2; ++dir) {
      k_conv_silu<<<cdiv(N_XC, TB), TB, 0, stream>>>(
          xz, convw + (size_t)i * DI * DCONV, convb + (size_t)i * DI, xc, xcbf, dir);
      // x_proj: [5632,256] x [48,256]^T -> dbc (N padded 40->48, one strip of 48)
      k_gemm_wmma<3><<<dim3(1, M / 64), gblk, 0, stream>>>(
          xcbf, DI, w_xp + (size_t)i * 48 * DI, DI, dbc, 48, nullptr, M, 48, DI);
      k_dt<<<cdiv(N_XC, TB), TB, 0, stream>>>(
          dbc, dtw + (size_t)i * DI * DTR, dtb + (size_t)i * DI, delta);
      const float* Ai = (dir == 0 ? negA : negAb) + (size_t)i * DI * DSTATE;
      k_scan<<<cdiv(BDIM * DI, TB), TB, 0, stream>>>(
          dbc, delta, xc, xz, Ai, Dparam + (size_t)i * DI, ybuf, dir);
    }

    { int n = N_XC; k_f2bf<<<cdiv(n, TB), TB, 0, stream>>>(ybuf, ybf, n); }
    // out_proj: [5632,256] x [128,256]^T -> hidden (N=128 -> 2 strips of 64)
    k_gemm_wmma<4><<<dim3(DM / 64, M / 64), gblk, 0, stream>>>(
        ybf, DI, w_op + (size_t)i * DM * DI, DI, hid, DM, nullptr, M, DM, DI);
    k_add<<<cdiv(N_RES, TB), TB, 0, stream>>>(res, hid, N_RES);
  }

  // final LN (f32 out, reuse hid), mean over tokens, classifier head
  k_ln<<<M, DM, 0, stream>>>(res, nfw, nfb, (__bf16*)nullptr, hid);
  k_meanT<<<cdiv(BDIM * DM, TB), TB, 0, stream>>>(hid, meanv);
  k_head<<<cdiv(BDIM * NCLS, TB), TB, 0, stream>>>(meanv, headw, headb, out);
}
